// Network_23922967839459
// MI455X (gfx1250) — compile-verified, matrix operand order checked by reading, not run on hardware
//
#include <hip/hip_runtime.h>
#include <hip/hip_bf16.h>

// ---------------- problem constants (match reference) ----------------
#define T_STEPS 20
#define M_PAD   32            // pad T to 2 WMMA M-tiles
#define N1      65536
#define N2      4096
#define KSPLIT  8             // K partitions -> 2048 independent GEMM waves
#define KCHUNK  (N1 / KSPLIT) // 8192

#define BETA            0.9f
#define FREQ_BETA       0.95f
#define TARGET_FREQ     0.1f
#define REFRACTORY_GAIN (-0.3f)
#define BASE_THRESHOLD  1.0f

typedef __attribute__((ext_vector_type(16))) __bf16 bf16x16;
typedef __attribute__((ext_vector_type(8)))  float  v8f;

// ---------------------------------------------------------------------
// Shared ensemble step (exact order of ops from reference)
// ---------------------------------------------------------------------
__device__ __forceinline__ float ensemble_step(float x, float& act, float& gain,
                                               float& thr, float& freq) {
    gain = gain + (1.0f - gain) * 0.2f;
    act  = BETA * act + x * gain + 0.05f;
    bool  spike = act > thr;
    float sf    = spike ? 1.0f : 0.0f;
    freq = FREQ_BETA * freq + (1.0f - FREQ_BETA) * sf;
    if (freq > TARGET_FREQ)      thr = thr + 0.05f;
    else if (freq < TARGET_FREQ) thr = thr / 1.05f;
    gain = spike ? REFRACTORY_GAIN : gain;
    act  = spike ? 0.0f : act;
    return sf;
}

// ---------------------------------------------------------------------
// Kernel 1: layer-1 dynamics for all T steps; emit bf16 spike matrix
// S is [M_PAD][N1] row-major, rows T_STEPS..M_PAD-1 zeroed (M padding).
// ---------------------------------------------------------------------
__global__ void layer1_spikes(const float* __restrict__ x_seq,
                              __bf16* __restrict__ S) {
    int i = blockIdx.x * blockDim.x + threadIdx.x;
    if (i >= N1) return;
    float act = 0.0f, gain = 1.0f, thr = BASE_THRESHOLD, freq = 0.0f;
#pragma unroll
    for (int t = 0; t < T_STEPS; ++t) {
        float sf = ensemble_step(x_seq[(size_t)t * N1 + i], act, gain, thr, freq);
        S[(size_t)t * N1 + i] = (__bf16)sf;
    }
#pragma unroll
    for (int t = T_STEPS; t < M_PAD; ++t)
        S[(size_t)t * N1 + i] = (__bf16)0.0f;
}

// ---------------------------------------------------------------------
// Kernel 2: partial GEMM  Ypart[ks][M_PAD][N2] = S[:, kchunk] @ W[kchunk, :]
// via v_wmma_f32_16x16x32_bf16. One wave per (ntile, kchunk); each wave
// computes BOTH M-tiles so every B fragment (and thus W) is loaded once.
//
// ISA lane layouts (CDNA5 ISA):
//   16-bit A 16x32: lane(0-15)=row m, elem j: K = (j/8)*16 + h*8 + (j%8),
//   h = lane/16. B (32x16) mirrored: col = lane%16, same per-elem K map.
//   f32 C/D: VGPR v, lane l: M = (l/16)*8 + v, N = l%16.
// ---------------------------------------------------------------------
__global__ void spike_gemm_wmma(const __bf16* __restrict__ S,
                                const float*  __restrict__ W,
                                float* __restrict__ Ypart) {
    const int wavesPerBlock = blockDim.x >> 5;
    const int wave  = blockIdx.x * wavesPerBlock + (threadIdx.x >> 5);
    const int lane  = threadIdx.x & 31;
    const int h     = lane >> 4;      // lane half
    const int idx   = lane & 15;      // row (A) / col (B) within tile

    const int nTilesN = N2 / 16;                 // 256
    const int ntile   = wave % nTilesN;
    const int ks      = wave / nTilesN;          // 0..KSPLIT-1
    if (ks >= KSPLIT) return;

    const int kbegin = ks * KCHUNK;
    const int ncol   = ntile * 16 + idx;         // B col this lane owns

    const __bf16* __restrict__ Srow0 = S + (size_t)idx        * N1;  // M-tile 0
    const __bf16* __restrict__ Srow1 = S + (size_t)(16 + idx) * N1;  // M-tile 1
    const float*  __restrict__ Wcol  = W + (size_t)ncol;

    v8f c0 = {};
    v8f c1 = {};

    union AFrag { bf16x16 v; uint4 q[2]; };

    for (int kk = kbegin; kk < kbegin + KCHUNK; kk += 32) {
        // prefetch the strided B stream one iteration ahead (global_prefetch_b8)
        __builtin_prefetch(&Wcol[(size_t)(kk + 32 + h * 8) * N2], 0, 1);

        // ---- A fragments: two contiguous 16B chunks per lane per tile ----
        AFrag a0, a1;
        a0.q[0] = *(const uint4*)(Srow0 + kk + h * 8);       // elems 0..7  -> K = h*8+0..7
        a0.q[1] = *(const uint4*)(Srow0 + kk + 16 + h * 8);  // elems 8..15 -> K = 16+h*8+0..7
        a1.q[0] = *(const uint4*)(Srow1 + kk + h * 8);
        a1.q[1] = *(const uint4*)(Srow1 + kk + 16 + h * 8);

        // ---- B fragment: per-lane column gather + f32->bf16 convert ----
        // Per load instruction the half-waves touch two contiguous 64B row
        // segments -> minimal-traffic coalescing without LDS staging.
        bf16x16 b;
#pragma unroll
        for (int j = 0; j < 16; ++j) {
            int k = kk + ((j >> 3) << 4) + (h << 3) + (j & 7);
            b[j] = (__bf16)Wcol[(size_t)k * N2];
        }

        c0 = __builtin_amdgcn_wmma_f32_16x16x32_bf16(
                 false, a0.v, false, b, (short)0, c0, false, false);
        c1 = __builtin_amdgcn_wmma_f32_16x16x32_bf16(
                 false, a1.v, false, b, (short)0, c1, false, false);
    }

    // ---- store both 16x16 D tiles unconditionally (no branchy tail) ----
    float* __restrict__ Yp = Ypart + (size_t)ks * M_PAD * N2;
#pragma unroll
    for (int v = 0; v < 8; ++v) {
        Yp[(size_t)(h * 8 + v)      * N2 + ncol] = c0[v];
        Yp[(size_t)(16 + h * 8 + v) * N2 + ncol] = c1[v];
    }
}

// ---------------------------------------------------------------------
// Kernel 3: deterministic K-split reduction -> Y[T_STEPS][N2]
// ---------------------------------------------------------------------
__global__ void reduce_ksplit(const float* __restrict__ Ypart,
                              float* __restrict__ Y) {
    int i = blockIdx.x * blockDim.x + threadIdx.x;   // t*N2 + n, t < T_STEPS
    if (i >= T_STEPS * N2) return;
    float s = 0.0f;
#pragma unroll
    for (int ks = 0; ks < KSPLIT; ++ks)
        s += Ypart[(size_t)ks * M_PAD * N2 + i];
    Y[i] = s;
}

// ---------------------------------------------------------------------
// Kernel 4: layer-2 dynamics over precomputed y vectors; record the
// post-reset activation (reference returns s2[0] each step).
// ---------------------------------------------------------------------
__global__ void layer2_dynamics(const float* __restrict__ Y,
                                float* __restrict__ out) {
    int n = blockIdx.x * blockDim.x + threadIdx.x;
    if (n >= N2) return;
    float act = 0.0f, gain = 1.0f, thr = BASE_THRESHOLD, freq = 0.0f;
#pragma unroll
    for (int t = 0; t < T_STEPS; ++t) {
        (void)ensemble_step(Y[(size_t)t * N2 + n], act, gain, thr, freq);
        out[(size_t)t * N2 + n] = act;   // post-reset activation
    }
}

// ---------------------------------------------------------------------
extern "C" void kernel_launch(void* const* d_in, const int* in_sizes, int n_in,
                              void* d_out, int out_size, void* d_ws, size_t ws_size,
                              hipStream_t stream) {
    const float* x_seq = (const float*)d_in[0];   // [T, N1]
    const float* W     = (const float*)d_in[1];   // [N1, N2]
    float*       out   = (float*)d_out;           // [T, N2]

    char* ws = (char*)d_ws;
    __bf16* S     = (__bf16*)ws;                                   // 4 MiB
    float*  Ypart = (float*)(ws + (size_t)M_PAD * N1 * sizeof(__bf16));        // 4 MiB
    float*  Y     = (float*)(ws + (size_t)M_PAD * N1 * sizeof(__bf16)
                                + (size_t)KSPLIT * M_PAD * N2 * sizeof(float)); // 320 KiB

    // Kernel 1: 65536 neurons
    layer1_spikes<<<N1 / 256, 256, 0, stream>>>(x_seq, S);

    // Kernel 2: 256 N-tiles x 8 K-chunks = 2048 waves, 8 waves/block
    const int totalWaves = (N2 / 16) * KSPLIT;
    spike_gemm_wmma<<<totalWaves / 8, 256, 0, stream>>>(S, W, Ypart);

    // Kernel 3: reduce K-split partials
    reduce_ksplit<<<(T_STEPS * N2) / 256, 256, 0, stream>>>(Ypart, Y);

    // Kernel 4: 4096 neurons
    layer2_dynamics<<<N2 / 256, 256, 0, stream>>>(Y, out);
}
